// Attention_53807350284519
// MI455X (gfx1250) — compile-verified, matrix operand order checked by reading, not run on hardware
//
#include <hip/hip_runtime.h>

typedef __attribute__((ext_vector_type(16))) __bf16 v16bf;
typedef __attribute__((ext_vector_type(8)))  float  v8f;
typedef __attribute__((ext_vector_type(4)))  unsigned int u32x4;
typedef __attribute__((ext_vector_type(8)))  int    i32x8;
typedef __attribute__((ext_vector_type(4)))  int    i32x4;

#define D_MODEL 1024
#define NHEAD   16
#define HD      64
#define SEQ     2048
#define BATCH   4
#define MTOT    (BATCH*SEQ)   // 8192

#define BM 128
#define BN 256
#define BK 32
#define LDST 40               // padded LDS row stride (bf16 elems): conflict-free, 16B aligned

#if defined(__gfx1250__) && defined(__has_builtin)
#if __has_builtin(__builtin_amdgcn_tensor_load_to_lds) && __has_builtin(__builtin_amdgcn_s_wait_tensorcnt)
#define USE_TDM 1
#endif
#endif

__device__ __forceinline__ unsigned short f2bf(float f) {
    unsigned int u = __float_as_uint(f);
    u += 0x7FFFu + ((u >> 16) & 1u);          // round-to-nearest-even
    return (unsigned short)(u >> 16);
}

struct Frag { union { v16bf v; uint4 q[2]; }; };

// ---------------- conversion kernels ----------------

__global__ void k_cvt(const float* __restrict__ in, unsigned short* __restrict__ out, int n) {
    int i = (blockIdx.x * blockDim.x + threadIdx.x) * 4;
    if (i + 3 < n) {
        float4 f = *(const float4*)(in + i);
        out[i + 0] = f2bf(f.x);
        out[i + 1] = f2bf(f.y);
        out[i + 2] = f2bf(f.z);
        out[i + 3] = f2bf(f.w);
    }
}

// Wt[n*K + k] = bf16(W[k*N + n])   (store weights transposed, N x K row-major)
__global__ void k_cvt_t(const float* __restrict__ W, unsigned short* __restrict__ Wt, int K, int N) {
    int idx = blockIdx.x * blockDim.x + threadIdx.x;
    if (idx >= N * K) return;
    int n = idx / K, k = idx - n * K;
    Wt[idx] = f2bf(W[(size_t)k * N + n]);
}

#ifdef USE_TDM
// Loop-invariant part of a 2-D TDM descriptor: tileRows x 32 bf16 tile out of a
// tensor with rowLen-elem rows, HW-padded in LDS: 16 dwords data + 4 dwords pad
// -> 40-elem (80 B) rows, matching LDST.
__device__ __forceinline__ i32x8 tdm_group1(int rowLen, int tileRows) {
    i32x8 g1;
    g1[0] = (1 << 16)                                     // data_size = 2 bytes
          | (1 << 20)                                     // pad_enable
          | (3 << 22)                                     // pad_interval code 3 = 16 dwords
          | (3 << 25);                                    // pad_amount  code 3 = 4 dwords
    g1[1] = (rowLen & 0xFFFF) << 16;                      // tensor_dim0[15:0]
    g1[2] = ((rowLen >> 16) & 0xFFFF)                     // tensor_dim0[31:16]
          | ((tileRows & 0xFFFF) << 16);                  // tensor_dim1[15:0]
    g1[3] = ((tileRows >> 16) & 0xFFFF)                   // tensor_dim1[31:16]
          | (BK << 16);                                   // tile_dim0 = 32
    g1[4] = tileRows & 0xFFFF;                            // tile_dim1
    g1[5] = rowLen;                                       // tensor_dim0_stride[31:0]
    g1[6] = 0;
    g1[7] = 0;
    return g1;
}

__device__ __forceinline__ void tdm_issue(const unsigned short* gtile,
                                          unsigned lds_byte_addr, i32x8 g1) {
    unsigned long long ga = (unsigned long long)gtile;
    u32x4 g0;
    g0[0] = 1u;                                           // count=1 (valid), user mode
    g0[1] = lds_byte_addr;                                // lds_addr
    g0[2] = (unsigned)(ga & 0xFFFFFFFFu);                 // global_addr[31:0]
    g0[3] = (unsigned)((ga >> 32) & 0x1FFFFFFu)           // global_addr[56:32]
          | (2u << 30);                                   // type = 2 ("image")
    i32x4 gz4 = {0, 0, 0, 0};
    i32x8 gz8 = {0, 0, 0, 0, 0, 0, 0, 0};
    __builtin_amdgcn_tensor_load_to_lds(g0, g1, gz4, gz4, gz8, 0);
}
#endif

// ---------------- GEMM: C(MxN) f32 = A(MxK)bf16 @ Bt(NxK)bf16^T ----------------
// Double-buffered tile pipeline: TDM (or cooperative loads) stage tile k+1 while
// the 8 waves run 16 WMMAs each on tile k.
// MODE 0: QKV epilogue (scatter to Q,K row-major per head; V transposed per head)
// MODE 1: out-proj epilogue (add bias, store f32)

template<int MODE>
__global__ __launch_bounds__(256) void k_gemm(
    const unsigned short* __restrict__ A,
    const unsigned short* __restrict__ Bt,
    const float* __restrict__ bias,
    unsigned short* __restrict__ Qb,
    unsigned short* __restrict__ Kb,
    unsigned short* __restrict__ Vt,
    float* __restrict__ Cout,
    int K, int N)
{
    __shared__ alignas(16) unsigned short lA[2][BM * LDST];
    __shared__ alignas(16) unsigned short lB[2][BN * LDST];

    const int tid  = threadIdx.x;
    const int lane = tid & 31, wid = tid >> 5;
    const int ln = lane & 15,  hi  = lane >> 4;
    const int wr = wid >> 2,   wc  = wid & 3;           // 2 x 4 wave grid, 64x64 wave tile
    const int rowBase = blockIdx.y * BM;
    const int colBase = blockIdx.x * BN;

#ifdef USE_TDM
    const i32x8 g1A = tdm_group1(K, BM);
    const i32x8 g1B = tdm_group1(K, BN);
    const unsigned ldsA[2] = { (unsigned)(unsigned long long)(&lA[0][0]),
                               (unsigned)(unsigned long long)(&lA[1][0]) };
    const unsigned ldsB[2] = { (unsigned)(unsigned long long)(&lB[0][0]),
                               (unsigned)(unsigned long long)(&lB[1][0]) };
#endif

    auto stage = [&](int buf, int k0) {
#ifdef USE_TDM
        if (wid == 0) {
            tdm_issue(A  + (size_t)rowBase * K + k0, ldsA[buf], g1A);
            tdm_issue(Bt + (size_t)colBase * K + k0, ldsB[buf], g1B);
        }
#else
#pragma unroll
        for (int i = 0; i < 2; ++i) {           // A: 512 b128, 2 per thread
            int v = i * 256 + tid;
            int row = v >> 2, cs = (v & 3) * 8;
            *(uint4*)&lA[buf][row * LDST + cs] = *(const uint4*)&A[(size_t)(rowBase + row) * K + k0 + cs];
        }
#pragma unroll
        for (int i = 0; i < 4; ++i) {           // B: 1024 b128, 4 per thread
            int v = i * 256 + tid;
            int row = v >> 2, cs = (v & 3) * 8;
            *(uint4*)&lB[buf][row * LDST + cs] = *(const uint4*)&Bt[(size_t)(colBase + row) * K + k0 + cs];
        }
#endif
    };

    v8f acc[4][4];
    v8f vz = {0.f,0.f,0.f,0.f,0.f,0.f,0.f,0.f};
#pragma unroll
    for (int t = 0; t < 4; ++t)
#pragma unroll
        for (int u = 0; u < 4; ++u) acc[t][u] = vz;

    // prologue: fill buffer 0
    stage(0, 0);
#ifdef USE_TDM
    if (wid == 0) __builtin_amdgcn_s_wait_tensorcnt(0);
#endif
    __syncthreads();

    for (int k0 = 0; k0 < K; k0 += BK) {
        const int cur = (k0 >> 5) & 1;
        if (k0 + BK < K) stage(cur ^ 1, k0 + BK);   // overlap DMA with WMMA

        Frag fa[4], fb[4];
#pragma unroll
        for (int t = 0; t < 4; ++t)
#pragma unroll
            for (int g = 0; g < 2; ++g)
                fa[t].q[g] = *(const uint4*)&lA[cur][(wr * 64 + t * 16 + ln) * LDST + hi * 8 + g * 16];
#pragma unroll
        for (int u = 0; u < 4; ++u)
#pragma unroll
            for (int g = 0; g < 2; ++g)
                fb[u].q[g] = *(const uint4*)&lB[cur][(wc * 64 + u * 16 + ln) * LDST + hi * 8 + g * 16];

#pragma unroll
        for (int t = 0; t < 4; ++t)
#pragma unroll
            for (int u = 0; u < 4; ++u)
                acc[t][u] = __builtin_amdgcn_wmma_f32_16x16x32_bf16(
                    false, fa[t].v, false, fb[u].v, (short)0, acc[t][u], false, false);

#ifdef USE_TDM
        if (wid == 0) __builtin_amdgcn_s_wait_tensorcnt(0);
#endif
        __syncthreads();
    }

    // epilogue
#pragma unroll
    for (int t = 0; t < 4; ++t)
#pragma unroll
        for (int u = 0; u < 4; ++u)
#pragma unroll
            for (int r = 0; r < 8; ++r) {
                int gm = rowBase + wr * 64 + t * 16 + r + hi * 8;
                int gn = colBase + wc * 64 + u * 16 + ln;
                float val = acc[t][u][r];
                if (MODE == 0) {
                    int which = gn >> 10, d = gn & 1023;
                    int h = d >> 6, hd = d & 63;
                    int b = gm >> 11, tok = gm & 2047;
                    unsigned short bv = f2bf(val);
                    if (which == 0)
                        Qb[(((size_t)(b * NHEAD + h)) * SEQ + tok) * HD + hd] = bv;
                    else if (which == 1)
                        Kb[(((size_t)(b * NHEAD + h)) * SEQ + tok) * HD + hd] = bv;
                    else
                        Vt[(((size_t)(b * NHEAD + h)) * HD + hd) * SEQ + tok] = bv;
                } else {
                    Cout[(size_t)gm * N + gn] = val + bias[gn];
                }
            }
}

// ---------------- attention: per (b,h), 3-phase softmax, deferred lane reductions --
// Q,K per head: (SEQ x HD) bf16 row-major.  V per head: (HD x SEQ) bf16 (transposed).
// One wave handles 16 query rows; block = 4 waves = 64 queries.

__global__ __launch_bounds__(128) void k_attn(
    const unsigned short* __restrict__ Qb,
    const unsigned short* __restrict__ Kb,
    const unsigned short* __restrict__ Vt,
    unsigned short* __restrict__ Ob)
{
    __shared__ alignas(16) unsigned short Pld[4][16 * LDST];

    const int tid  = threadIdx.x;
    const int lane = tid & 31, wid = tid >> 5;
    const int ln = lane & 15,  hi  = lane >> 4;
    const int b = blockIdx.z, h = blockIdx.y;
    const int qbase = blockIdx.x * 64 + wid * 16;
    const float scale = 0.125f;                  // 1/sqrt(64)

    const size_t headOff = ((size_t)(b * NHEAD + h)) * SEQ * HD;
    const unsigned short* Qh = Qb + headOff;
    const unsigned short* Kh = Kb + headOff;
    const unsigned short* Vh = Vt + headOff;     // (HD x SEQ)

    // Q fragments (16 x 64), resident for the whole kernel
    Frag qf[2];
#pragma unroll
    for (int f = 0; f < 2; ++f)
#pragma unroll
        for (int g = 0; g < 2; ++g)
            qf[f].q[g] = *(const uint4*)&Qh[(size_t)(qbase + ln) * HD + f * 32 + hi * 8 + g * 16];

    v8f vz = {0.f,0.f,0.f,0.f,0.f,0.f,0.f,0.f};

    // ---- phase A: per-lane raw row max (no shuffles, no exp in the hot loop) ----
    float mrow[8];
#pragma unroll
    for (int r = 0; r < 8; ++r) mrow[r] = -3.0e38f;

    for (int kc = 0; kc < SEQ; kc += 32) {
        v8f s0 = vz, s1 = vz;
#pragma unroll
        for (int f = 0; f < 2; ++f) {
            Frag kf;
#pragma unroll
            for (int g = 0; g < 2; ++g)
                kf.q[g] = *(const uint4*)&Kh[(size_t)(kc + ln) * HD + f * 32 + hi * 8 + g * 16];
            s0 = __builtin_amdgcn_wmma_f32_16x16x32_bf16(false, qf[f].v, false, kf.v, (short)0, s0, false, false);
        }
#pragma unroll
        for (int f = 0; f < 2; ++f) {
            Frag kf;
#pragma unroll
            for (int g = 0; g < 2; ++g)
                kf.q[g] = *(const uint4*)&Kh[(size_t)(kc + 16 + ln) * HD + f * 32 + hi * 8 + g * 16];
            s1 = __builtin_amdgcn_wmma_f32_16x16x32_bf16(false, qf[f].v, false, kf.v, (short)0, s1, false, false);
        }
#pragma unroll
        for (int r = 0; r < 8; ++r)
            mrow[r] = fmaxf(mrow[r], fmaxf(s0[r], s1[r]));
    }
    // one cross-lane max reduction at the end (rows live in 16-lane groups)
#pragma unroll
    for (int r = 0; r < 8; ++r) {
#pragma unroll
        for (int m = 1; m <= 8; m <<= 1)
            mrow[r] = fmaxf(mrow[r], __shfl_xor(mrow[r], m, 32));
        mrow[r] *= scale;                        // max of scaled scores
    }

    // ---- phase B: O += exp(s*scale - m) @ V ; row-sums accumulated per lane ----
    v8f oacc[4];
#pragma unroll
    for (int u = 0; u < 4; ++u) oacc[u] = vz;
    float psum[8];
#pragma unroll
    for (int r = 0; r < 8; ++r) psum[r] = 0.f;

    unsigned short* Pw = &Pld[wid][0];

    for (int kc = 0; kc < SEQ; kc += 32) {
        v8f s0 = vz, s1 = vz;
#pragma unroll
        for (int f = 0; f < 2; ++f) {
            Frag kf;
#pragma unroll
            for (int g = 0; g < 2; ++g)
                kf.q[g] = *(const uint4*)&Kh[(size_t)(kc + ln) * HD + f * 32 + hi * 8 + g * 16];
            s0 = __builtin_amdgcn_wmma_f32_16x16x32_bf16(false, qf[f].v, false, kf.v, (short)0, s0, false, false);
        }
#pragma unroll
        for (int f = 0; f < 2; ++f) {
            Frag kf;
#pragma unroll
            for (int g = 0; g < 2; ++g)
                kf.q[g] = *(const uint4*)&Kh[(size_t)(kc + 16 + ln) * HD + f * 32 + hi * 8 + g * 16];
            s1 = __builtin_amdgcn_wmma_f32_16x16x32_bf16(false, qf[f].v, false, kf.v, (short)0, s1, false, false);
        }

        // exponentiate, accumulate per-lane row sums, C-layout -> A-layout via LDS
#pragma unroll
        for (int r = 0; r < 8; ++r) {
            int row = r + hi * 8;
            float p0 = __expf(s0[r] * scale - mrow[r]);
            float p1 = __expf(s1[r] * scale - mrow[r]);
            psum[r] += p0 + p1;
            Pw[row * LDST + ln]      = f2bf(p0);
            Pw[row * LDST + 16 + ln] = f2bf(p1);
        }
        asm volatile("s_wait_dscnt 0" ::: "memory");   // per-wave LDS RAW fence

        Frag pf;
        pf.q[0] = *(const uint4*)&Pw[ln * LDST + hi * 8];
        pf.q[1] = *(const uint4*)&Pw[ln * LDST + 16 + hi * 8];

#pragma unroll
        for (int u = 0; u < 4; ++u) {
            Frag vf;
#pragma unroll
            for (int g = 0; g < 2; ++g)
                vf.q[g] = *(const uint4*)&Vh[(size_t)(u * 16 + ln) * SEQ + kc + hi * 8 + g * 16];
            oacc[u] = __builtin_amdgcn_wmma_f32_16x16x32_bf16(false, pf.v, false, vf.v, (short)0, oacc[u], false, false);
        }
    }

    // one cross-lane sum reduction for the softmax denominators
    float lrow[8];
#pragma unroll
    for (int r = 0; r < 8; ++r) {
        float s = psum[r];
#pragma unroll
        for (int m = 1; m <= 8; m <<= 1) s += __shfl_xor(s, m, 32);
        lrow[r] = s;
    }

    // normalize + store bf16 into (B, N, D) layout for the out-projection GEMM
#pragma unroll
    for (int u = 0; u < 4; ++u)
#pragma unroll
        for (int r = 0; r < 8; ++r) {
            float o = oacc[u][r] / lrow[r];
            int gq = qbase + r + hi * 8;
            Ob[((size_t)(b * SEQ + gq)) * D_MODEL + h * HD + u * 16 + ln] = f2bf(o);
        }
}

// ---------------- launcher ----------------

extern "C" void kernel_launch(void* const* d_in, const int* in_sizes, int n_in,
                              void* d_out, int out_size, void* d_ws, size_t ws_size,
                              hipStream_t stream) {
    (void)in_sizes; (void)n_in; (void)out_size; (void)ws_size;
    const float* x      = (const float*)d_in[0];
    const float* W_qkv  = (const float*)d_in[1];
    const float* W_proj = (const float*)d_in[2];
    const float* b_proj = (const float*)d_in[3];
    float* out = (float*)d_out;

    const size_t NX  = (size_t)MTOT * D_MODEL;      // 8,388,608
    unsigned short* Xb     = (unsigned short*)d_ws;
    unsigned short* WqkvT  = Xb     + NX;                            // 3072 x 1024
    unsigned short* WprojT = WqkvT  + (size_t)3 * D_MODEL * D_MODEL; // 1024 x 1024
    unsigned short* Qb     = WprojT + (size_t)D_MODEL * D_MODEL;     // B*H*SEQ*HD == NX
    unsigned short* Kb     = Qb + NX;
    unsigned short* Vt     = Kb + NX;
    unsigned short* Ob     = Vt + NX;

    // 1. convert inputs to bf16 (weights transposed to N x K)
    k_cvt  <<<(int)(NX / 1024), 256, 0, stream>>>(x, Xb, (int)NX);
    k_cvt_t<<<(3 * D_MODEL * D_MODEL) / 256, 256, 0, stream>>>(W_qkv,  WqkvT,  D_MODEL, 3 * D_MODEL);
    k_cvt_t<<<(D_MODEL * D_MODEL) / 256,     256, 0, stream>>>(W_proj, WprojT, D_MODEL, D_MODEL);

    // 2. QKV projection, scatter to per-head Q, K (row-major) and V^T
    k_gemm<0><<<dim3(3 * D_MODEL / BN, MTOT / BM), 256, 0, stream>>>(
        Xb, WqkvT, nullptr, Qb, Kb, Vt, nullptr, D_MODEL, 3 * D_MODEL);

    // 3. attention (3-phase streamed softmax, all WMMA)
    k_attn<<<dim3(SEQ / 64, NHEAD, BATCH), 128, 0, stream>>>(Qb, Kb, Vt, Ob);

    // 4. output projection + bias (f32 out)
    k_gemm<1><<<dim3(D_MODEL / BN, MTOT / BM), 256, 0, stream>>>(
        Ob, WprojT, b_proj, nullptr, nullptr, nullptr, out, D_MODEL, D_MODEL);
}
